// FF_27650999451977
// MI455X (gfx1250) — matrix operand order repaired and checked by reading, not compiled, not run on hardware
//
#include <hip/hip_runtime.h>
#include <hip/hip_bf16.h>

// ---------------------------------------------------------------------------
// 2-layer gated RNN scan on MI455X (gfx1250, wave32, WMMA).
//   Phase 0 (prep):  f32->f16 conversion of x and weights, combined biases,
//                    zero h state, reset grid barrier.
//   Phase 1 (xproj): bulk parallel WMMA GEMM of layer-0 input projections
//                    for ALL timesteps:  Xg0 = x @ Wgx[0]^T   (g in {j,k})
//   Phase 2 (rnn):   persistent kernel, 16 blocks x 256 threads, 512 steps,
//                    grid barrier between layer phases. Each wave owns one
//                    16x16 output tile and both gates (shared A fragment).
// ---------------------------------------------------------------------------

typedef __attribute__((ext_vector_type(16))) _Float16 v16h;
typedef __attribute__((ext_vector_type(8)))  _Float16 v8h;
typedef __attribute__((ext_vector_type(8)))  float    v8f;

#define SEQL 512
#define BB   64
#define HH   512
#define KDIM 512
#define LL   2
#define NWG  16
#define WMAT (HH * KDIM)   // 262144 elements per [H][K] matrix

union V16u { v16h v; v8h h[2]; };

// A fragment (16x32 f16, M x K): lane<16 -> rows M=lane, K in {kk..kk+15}
//                                lane>=16 -> same rows, K base +16.
__device__ __forceinline__ v16h load_afrag(const _Float16* __restrict__ base,
                                           int m, int kb) {
  V16u u;
  u.h[0] = *(const v8h*)(base + (long)m * KDIM + kb);
  u.h[1] = *(const v8h*)(base + (long)m * KDIM + kb + 8);
  return u.v;
}

// B fragment (32x16 f16, K x N): B[k][n] = W[n][k]  (torch weight is [out,in]),
// so lane's 16 elements are one contiguous 32-byte run of row n of W.
__device__ __forceinline__ v16h load_bfrag(const _Float16* __restrict__ w,
                                           int n, int kb) {
  return *(const v16h*)(w + (long)n * KDIM + kb);
}

__device__ __forceinline__ v8f wmma_f16(v16h a, v16h b, v8f c) {
  return __builtin_amdgcn_wmma_f32_16x16x32_f16(false, a, false, b,
                                                (short)0, c, false, false);
}

__device__ __forceinline__ float sigmoidf_(float x) {
  return 1.0f / (1.0f + __expf(-x));
}

// Sense-reversing grid barrier over NWG resident blocks (bar[0]=count, bar[1]=gen).
__device__ __forceinline__ void grid_barrier(unsigned* __restrict__ bar) {
  __syncthreads();
  if (threadIdx.x == 0) {
    __threadfence();
    unsigned gen  = __hip_atomic_load(&bar[1], __ATOMIC_ACQUIRE, __HIP_MEMORY_SCOPE_AGENT);
    unsigned prev = __hip_atomic_fetch_add(&bar[0], 1u, __ATOMIC_ACQ_REL, __HIP_MEMORY_SCOPE_AGENT);
    if (prev == NWG - 1u) {
      __hip_atomic_store(&bar[0], 0u, __ATOMIC_RELAXED, __HIP_MEMORY_SCOPE_AGENT);
      __hip_atomic_fetch_add(&bar[1], 1u, __ATOMIC_RELEASE, __HIP_MEMORY_SCOPE_AGENT);
    } else {
      while (__hip_atomic_load(&bar[1], __ATOMIC_ACQUIRE, __HIP_MEMORY_SCOPE_AGENT) == gen)
        __builtin_amdgcn_s_sleep(1);
    }
  }
  __syncthreads();
}

// ---------------------------------------------------------------------------
__global__ void prep_kernel(const float* __restrict__ x,
                            const float* __restrict__ Wjx, const float* __restrict__ bjx,
                            const float* __restrict__ Wjh, const float* __restrict__ bjh,
                            const float* __restrict__ Wkx, const float* __restrict__ bkx,
                            const float* __restrict__ Wkh, const float* __restrict__ bkh,
                            _Float16* __restrict__ x16, _Float16* __restrict__ w16,
                            float* __restrict__ bsum,
                            float* __restrict__ hbuf, _Float16* __restrict__ h16,
                            unsigned* __restrict__ bar) {
  const long i0 = (long)blockIdx.x * blockDim.x + threadIdx.x;
  const long stride = (long)gridDim.x * blockDim.x;
  const long NX = (long)SEQL * BB * KDIM;   // x elements
  const long NW = (long)4 * LL * WMAT;      // all 4 weight tensors
  const long NB = (long)2 * LL * HH;        // combined biases [gate][l][n]
  const long NH = (long)2 * LL * BB * HH;   // double-buffered h state

  for (long i = i0; i < NX; i += stride) x16[i] = (_Float16)x[i];

  for (long i = i0; i < NW; i += stride) {
    long t = i / (LL * WMAT), r = i % (LL * WMAT);
    const float* src = (t == 0) ? Wjx : (t == 1) ? Wjh : (t == 2) ? Wkx : Wkh;
    w16[i] = (_Float16)src[r];
  }
  for (long i = i0; i < NB; i += stride) {
    long g = i / (LL * HH), r = i % (LL * HH);
    bsum[i] = (g == 0) ? (bjx[r] + bjh[r]) : (bkx[r] + bkh[r]);
  }
  for (long i = i0; i < NH; i += stride) { hbuf[i] = 0.0f; h16[i] = (_Float16)0.0f; }
  if (i0 == 0) { bar[0] = 0u; bar[1] = 0u; }
}

// ---------------------------------------------------------------------------
// Phase 1: Xg0[t*B+m, n] = sum_k x16[t*B+m, k] * Wgx0[n, k]   (no bias here)
// 2 gates x 2048 m-tiles x 32 n-tiles = 131072 waves -> 16384 blocks of 8 waves.
__global__ void __launch_bounds__(256)
xproj_kernel(const _Float16* __restrict__ x16, const _Float16* __restrict__ w16,
             _Float16* __restrict__ xj0, _Float16* __restrict__ xk0) {
  const int wid = threadIdx.x >> 5, lane = threadIdx.x & 31;
  const int tid = blockIdx.x * 8 + wid;
  const int g   = tid >> 16;          // 0 = j, 1 = k
  const int rem = tid & 65535;
  const int mt  = rem >> 5;
  const int nt  = rem & 31;

  const _Float16* Wg = w16 + (g ? 4 : 0) * (long)LL * WMAT;  // Wjx[0] / Wkx[0]
  const int mA = mt * 16 + (lane & 15);
  const int nB = nt * 16 + (lane & 15);
  const int kbA = (lane < 16) ? 0 : 16;
  const int kbB = (lane < 16) ? 0 : 16;

  v8f acc = {};
  for (int kk = 0; kk < KDIM; kk += 32) {
    v16h a = load_afrag(x16, mA, kk + kbA);
    v16h b = load_bfrag(Wg, nB, kk + kbB);
    acc = wmma_f16(a, b, acc);
  }
  _Float16* dst = g ? xk0 : xj0;
  const int m0 = mt * 16 + ((lane < 16) ? 0 : 8);
#pragma unroll
  for (int r = 0; r < 8; ++r)
    dst[(long)(m0 + r) * HH + nB] = (_Float16)acc[r];
}

// ---------------------------------------------------------------------------
// Phase 2: persistent scan. 16 blocks x 8 waves. Wave -> one 16x16 tile:
//   m0 = 16*(wid&3), n0 = 32*blockIdx.x + 16*(wid>>2); computes j AND k gates.
__global__ void __launch_bounds__(256, 1)
rnn_kernel(const _Float16* __restrict__ w16, const float* __restrict__ bsum,
           const _Float16* __restrict__ xj0, const _Float16* __restrict__ xk0,
           float* __restrict__ hbuf, _Float16* __restrict__ h16,
           float* __restrict__ out, unsigned* __restrict__ bar) {
  const int wid  = threadIdx.x >> 5, lane = threadIdx.x & 31;
  const int m0   = (wid & 3) * 16;
  const int n0   = blockIdx.x * 32 + (wid >> 2) * 16;
  const int lhi  = (lane < 16) ? 0 : 1;
  const int nlane  = n0 + (lane & 15);         // N coord (B frag & C/D cols)
  const int mlaneA = m0 + (lane & 15);         // M row for A frags
  const int kbA = lhi ? 16 : 0;
  const int kbB = lhi ? 16 : 0;
  const int mrow0 = m0 + (lhi ? 8 : 0);        // first M row of C/D per lane

  for (int t = 0; t < SEQL; ++t) {
    const int p = t & 1;                        // read buffer parity
    for (int l = 0; l < LL; ++l) {
      const float bj = bsum[l * HH + nlane];
      const float bk = bsum[LL * HH + l * HH + nlane];
      v8f accj = {bj, bj, bj, bj, bj, bj, bj, bj};
      v8f acck = {bk, bk, bk, bk, bk, bk, bk, bk};

      // hidden-state path: h_cur[l] @ Wgh[l]^T
      const _Float16* hA  = h16 + (long)(p * 2 + l) * (BB * HH);
      const _Float16* wjh = w16 + (long)(2 + l) * WMAT;
      const _Float16* wkh = w16 + (long)(6 + l) * WMAT;
      for (int kk = 0; kk < KDIM; kk += 32) {
        v16h a   = load_afrag(hA, mlaneA, kk + kbA);
        v16h bjf = load_bfrag(wjh, nlane, kk + kbB);
        v16h bkf = load_bfrag(wkh, nlane, kk + kbB);
        accj = wmma_f16(a, bjf, accj);
        acck = wmma_f16(a, bkf, acck);
      }

      if (l == 0) {
        // input path was precomputed in parallel for all timesteps
        const _Float16* xj = xj0 + (long)t * BB * HH;
        const _Float16* xk = xk0 + (long)t * BB * HH;
        if (t + 1 < SEQL) {  // global_prefetch_b8 next timestep's tiles
          __builtin_prefetch(xj + (long)BB * HH + (long)mrow0 * HH + nlane, 0, 1);
          __builtin_prefetch(xk + (long)BB * HH + (long)mrow0 * HH + nlane, 0, 1);
        }
#pragma unroll
        for (int r = 0; r < 8; ++r) {
          long idx = (long)(mrow0 + r) * HH + nlane;
          accj[r] += (float)xj[idx];
          acck[r] += (float)xk[idx];
        }
      } else {
        // input path: this timestep's layer-0 output @ Wgx[1]^T
        const _Float16* iA  = h16 + (long)((1 - p) * 2 + 0) * (BB * HH);
        const _Float16* wjx = w16 + (long)(0 + l) * WMAT;
        const _Float16* wkx = w16 + (long)(4 + l) * WMAT;
        for (int kk = 0; kk < KDIM; kk += 32) {
          v16h a   = load_afrag(iA, mlaneA, kk + kbA);
          v16h bjf = load_bfrag(wjx, nlane, kk + kbB);
          v16h bkf = load_bfrag(wkx, nlane, kk + kbB);
          accj = wmma_f16(a, bjf, accj);
          acck = wmma_f16(a, bkf, acck);
        }
      }

      // epilogue: gates + blend, write f32 state, f16 WMMA copy, outputs
      const float*    hold   = hbuf + (long)(p * 2 + l) * (BB * HH);
      float*          hnew   = hbuf + (long)((1 - p) * 2 + l) * (BB * HH);
      _Float16*       hnew16 = h16  + (long)((1 - p) * 2 + l) * (BB * HH);
#pragma unroll
      for (int r = 0; r < 8; ++r) {
        const int m = mrow0 + r;
        const long idx = (long)m * HH + nlane;
        float j  = sigmoidf_(accj[r]);
        float k  = sigmoidf_(acck[r]);
        float ho = hold[idx];
        float hn = j * (1.0f - ho) + (1.0f - k) * ho;
        hnew[idx]   = hn;
        hnew16[idx] = (_Float16)hn;
        if (l == LL - 1)
          out[((long)t * BB + m) * HH + nlane] = hn;          // outputs[t]
        if (t == SEQL - 1)
          out[(long)SEQL * BB * HH + ((long)l * BB + m) * HH + nlane] = hn;  // hidden
      }
      grid_barrier(bar);  // layer-0 output / new state visible to all blocks
    }
  }
}

// ---------------------------------------------------------------------------
extern "C" void kernel_launch(void* const* d_in, const int* in_sizes, int n_in,
                              void* d_out, int out_size, void* d_ws, size_t ws_size,
                              hipStream_t stream) {
  (void)in_sizes; (void)n_in; (void)out_size; (void)ws_size;
  const float* x   = (const float*)d_in[0];
  const float* Wjx = (const float*)d_in[1];
  const float* bjx = (const float*)d_in[2];
  const float* Wjh = (const float*)d_in[3];
  const float* bjh = (const float*)d_in[4];
  const float* Wkx = (const float*)d_in[5];
  const float* bkx = (const float*)d_in[6];
  const float* Wkh = (const float*)d_in[7];
  const float* bkh = (const float*)d_in[8];

  char* ws = (char*)d_ws;
  size_t off = 0;
  unsigned* bar = (unsigned*)(ws + off); off += 256;
  _Float16* x16 = (_Float16*)(ws + off); off += (size_t)SEQL * BB * KDIM * 2;  // 32 MB
  _Float16* w16 = (_Float16*)(ws + off); off += (size_t)4 * LL * WMAT * 2;     //  4 MB
  float*   bsum = (float*)   (ws + off); off += (size_t)2 * LL * HH * 4;
  _Float16* xj0 = (_Float16*)(ws + off); off += (size_t)SEQL * BB * HH * 2;    // 32 MB
  _Float16* xk0 = (_Float16*)(ws + off); off += (size_t)SEQL * BB * HH * 2;    // 32 MB
  float*   hbuf = (float*)   (ws + off); off += (size_t)2 * LL * BB * HH * 4;
  _Float16* h16 = (_Float16*)(ws + off); off += (size_t)2 * LL * BB * HH * 2;

  prep_kernel<<<8192, 256, 0, stream>>>(x, Wjx, bjx, Wjh, bjh, Wkx, bkx, Wkh, bkh,
                                        x16, w16, bsum, hbuf, h16, bar);
  xproj_kernel<<<16384, 256, 0, stream>>>(x16, w16, xj0, xk0);
  rnn_kernel<<<NWG, 256, 0, stream>>>(w16, bsum, xj0, xk0, hbuf, h16,
                                      (float*)d_out, bar);
}
